// MultiHeadSelfAttention_42236708388912
// MI455X (gfx1250) — matrix operand order repaired and checked by reading, not compiled, not run on hardware
//
#include <hip/hip_runtime.h>

#define HIDW 768
#define BATCH 4
#define SEQ 2048
#define NH 12
#define HD 64

typedef __attribute__((ext_vector_type(16))) __bf16 v16bf;
typedef __attribute__((ext_vector_type(8)))  __bf16 v8bf;
typedef __attribute__((ext_vector_type(8)))  float  v8f;

#define WMMA_BF16(A, Bf, C) \
  __builtin_amdgcn_wmma_f32_16x16x32_bf16(false, (A), false, (Bf), (short)0, (C), false, false)

// ---------------- WMMA fragment loaders (match CDNA5 ISA VGPR layouts) ----------------

// A (16x32 bf16, row-major, stride lda elems): lane row = lane&15, half = lane>>4.
// Elements e<8 -> K = e + 8*half ; e>=8 -> K = e + 8 + 8*half  (two 16B chunks).
__device__ __forceinline__ v16bf loadA_bf16(const __bf16* base, int lda, int lane) {
  const int row = lane & 15, half = lane >> 4;
  const __bf16* p = base + row * lda + half * 8;
  v8bf lo = *(const v8bf*)(p);
  v8bf hi = *(const v8bf*)(p + 16);
  return __builtin_shufflevector(lo, hi, 0,1,2,3,4,5,6,7,8,9,10,11,12,13,14,15);
}

// Same A fragment but from f32 memory with on-the-fly bf16 conversion.
__device__ __forceinline__ v16bf loadA_f32(const float* base, int lda, int lane) {
  const int row = lane & 15, half = lane >> 4;
  const float* p = base + row * lda + half * 8;
  const float4 c0 = *(const float4*)(p);
  const float4 c1 = *(const float4*)(p + 4);
  const float4 c2 = *(const float4*)(p + 16);
  const float4 c3 = *(const float4*)(p + 20);
  v16bf a;
  a[0]=(__bf16)c0.x;  a[1]=(__bf16)c0.y;  a[2]=(__bf16)c0.z;  a[3]=(__bf16)c0.w;
  a[4]=(__bf16)c1.x;  a[5]=(__bf16)c1.y;  a[6]=(__bf16)c1.z;  a[7]=(__bf16)c1.w;
  a[8]=(__bf16)c2.x;  a[9]=(__bf16)c2.y;  a[10]=(__bf16)c2.z; a[11]=(__bf16)c2.w;
  a[12]=(__bf16)c3.x; a[13]=(__bf16)c3.y; a[14]=(__bf16)c3.z; a[15]=(__bf16)c3.w;
  return a;
}

// B (32x16 bf16): lane col = lane&15, e<8 -> K = e + 8*half ; e>=8 -> K = e + 8 + 8*half. Memory layout must have the
// contraction dim contiguous per column: element B[k][n] at base + n*ldb + k.
__device__ __forceinline__ v16bf loadB_k16(const __bf16* base, int ldb, int lane) {
  const int col = lane & 15, half = lane >> 4;
  const __bf16* p = base + col * ldb + half * 8;
  v8bf lo = *(const v8bf*)(p);
  v8bf hi = *(const v8bf*)(p + 16);
  return __builtin_shufflevector(lo, hi, 0,1,2,3,4,5,6,7,8,9,10,11,12,13,14,15);
}

// ---------------- Kernel 1: transpose + f32->bf16 convert the 4 weight matrices ----------------
__global__ void w_transpose_kernel(const float* __restrict__ W0, const float* __restrict__ W1,
                                   const float* __restrict__ W2, const float* __restrict__ W3,
                                   __bf16* __restrict__ WTbase) {
  const float* W = (blockIdx.z == 0) ? W0 : (blockIdx.z == 1) ? W1 : (blockIdx.z == 2) ? W2 : W3;
  __bf16* WT = WTbase + (size_t)blockIdx.z * HIDW * HIDW;
  __shared__ __bf16 tile[32][33];
  const int bx = blockIdx.x * 32;  // n base
  const int by = blockIdx.y * 32;  // k base
  const int tx = threadIdx.x, ty = threadIdx.y;
  #pragma unroll
  for (int i = ty; i < 32; i += 8)
    tile[i][tx] = (__bf16)W[(size_t)(by + i) * HIDW + (bx + tx)];
  __syncthreads();
  #pragma unroll
  for (int i = ty; i < 32; i += 8)
    WT[(size_t)(bx + i) * HIDW + (by + tx)] = tile[tx][i];
}

// ---------------- Kernel 2: fused QKV projection ----------------
// Each wave: one 16x64 output tile of one of {Q,K,V}. Q,K -> [B,H,S,64] bf16; V -> [B,H,64,S] bf16.
__global__ void qkv_gemm_kernel(const float* __restrict__ x,
                                const __bf16* __restrict__ WqT, const __bf16* __restrict__ WkT,
                                const __bf16* __restrict__ WvT,
                                const float* __restrict__ bq, const float* __restrict__ bk,
                                const float* __restrict__ bv,
                                __bf16* __restrict__ Q, __bf16* __restrict__ K,
                                __bf16* __restrict__ Vt) {
  const int wid  = (blockIdx.x * blockDim.x + threadIdx.x) >> 5;
  const int lane = threadIdx.x & 31;
  const int MT = (BATCH * SEQ) / 16;  // 512
  const int NG = HIDW / 64;           // 12
  const int mt  = wid % MT;
  const int t   = wid / MT;
  const int ng  = t % NG;
  const int mat = t / NG;  // 0=Q 1=K 2=V
  const __bf16* WT   = (mat == 0) ? WqT : (mat == 1) ? WkT : WvT;
  const float*  bias = (mat == 0) ? bq  : (mat == 1) ? bk  : bv;

  const float* Abase = x + (size_t)mt * 16 * HIDW;
  v8f acc[4] = {v8f{0}, v8f{0}, v8f{0}, v8f{0}};
  for (int kk = 0; kk < HIDW; kk += 32) {
    v16bf a = loadA_f32(Abase + kk, HIDW, lane);
    #pragma unroll
    for (int j = 0; j < 4; ++j) {
      v16bf bf = loadB_k16(WT + (size_t)(ng * 64 + j * 16) * HIDW + kk, HIDW, lane);
      acc[j] = WMMA_BF16(a, bf, acc[j]);
    }
  }
  const int row0 = mt * 16 + ((lane >> 4) << 3);
  #pragma unroll
  for (int j = 0; j < 4; ++j) {
    const int n = ng * 64 + j * 16 + (lane & 15);
    const float bb = bias[n];
    const int h = n >> 6, d = n & 63;
    #pragma unroll
    for (int r = 0; r < 8; ++r) {
      const int m = row0 + r;
      const int b = m >> 11, s = m & 2047;
      const __bf16 e = (__bf16)(acc[j][r] + bb);
      if (mat == 0)      Q [((size_t)(b * NH + h) * SEQ + s) * HD + d] = e;
      else if (mat == 1) K [((size_t)(b * NH + h) * SEQ + s) * HD + d] = e;
      else               Vt[((size_t)(b * NH + h) * HD  + d) * SEQ + s] = e;
    }
  }
}

// ---------------- Kernel 3: fused scores + softmax + attn-write + P@V ----------------
// One 256-thread block (8 waves) per (b,h, 16-row stripe).
//  wave w: scores for cols [w*256, w*256+256)  (16 WMMA n-tiles x 2 k-steps)
//  softmax: in-register + shfl_xor within wave, cross-wave via small LDS buffer
//  probabilities: ONE nontemporal f32 write to global attn + bf16 copy in LDS stripe
//  P@V: wave w contracts its own 256 columns from LDS; partials reduced via LDS.
#define SP_PITCH 2064  // 2048 + 16 bf16 pad -> 4128B row pitch (bank decorrelation)

__global__ void attn_fused_kernel(const __bf16* __restrict__ Q, const __bf16* __restrict__ K,
                                  const __bf16* __restrict__ Vt,
                                  float* __restrict__ attn, __bf16* __restrict__ ctx) {
  __shared__ __bf16 sP[16 * SP_PITCH];   // 66,048 B probability stripe (bf16)
  __shared__ float  redbuf[16][8];       // cross-wave row reductions

  const int w    = threadIdx.x >> 5;     // wave id 0..7
  const int lane = threadIdx.x & 31;
  const int half = lane >> 4;
  const int l16  = lane & 15;
  const int mt = blockIdx.x & 127;       // 16-row stripe within sequence
  const int bh = blockIdx.x >> 7;        // fused (batch, head) 0..47

  const __bf16* Qb = Q + (size_t)bh * SEQ * HD + (size_t)mt * 16 * HD;
  const __bf16* Kb = K + (size_t)bh * SEQ * HD + (size_t)(w * 256) * HD;

  // ---- phase 1: scores 16x256 in registers ----
  v8f accS[16];
  #pragma unroll
  for (int j = 0; j < 16; ++j) accS[j] = v8f{0};
  #pragma unroll
  for (int kk = 0; kk < HD; kk += 32) {
    v16bf a = loadA_bf16(Qb + kk, HD, lane);
    #pragma unroll
    for (int j = 0; j < 16; ++j) {
      v16bf bf = loadB_k16(Kb + (size_t)(j * 16) * HD + kk, HD, lane);
      accS[j] = WMMA_BF16(a, bf, accS[j]);
    }
  }
  #pragma unroll
  for (int j = 0; j < 16; ++j)
    #pragma unroll
    for (int r = 0; r < 8; ++r) accS[j][r] *= 0.125f;  // 1/sqrt(64)

  // ---- phase 2: softmax over full 2048-wide rows ----
  float rowmax[8], rowinv[8];
  #pragma unroll
  for (int r = 0; r < 8; ++r) {
    float m = accS[0][r];
    #pragma unroll
    for (int j = 1; j < 16; ++j) m = fmaxf(m, accS[j][r]);
    #pragma unroll
    for (int d = 1; d < 16; d <<= 1) m = fmaxf(m, __shfl_xor(m, d, 32));
    rowmax[r] = m;
  }
  if (l16 == 0) {
    #pragma unroll
    for (int r = 0; r < 8; ++r) redbuf[r + 8 * half][w] = rowmax[r];
  }
  __syncthreads();
  #pragma unroll
  for (int r = 0; r < 8; ++r) {
    float m = redbuf[r + 8 * half][0];
    #pragma unroll
    for (int ww = 1; ww < 8; ++ww) m = fmaxf(m, redbuf[r + 8 * half][ww]);
    rowmax[r] = m;
  }
  __syncthreads();  // redbuf reuse for sums
  #pragma unroll
  for (int r = 0; r < 8; ++r) {
    float s = 0.f;
    #pragma unroll
    for (int j = 0; j < 16; ++j) {
      const float e = __expf(accS[j][r] - rowmax[r]);
      accS[j][r] = e;
      s += e;
    }
    #pragma unroll
    for (int d = 1; d < 16; d <<= 1) s += __shfl_xor(s, d, 32);
    rowinv[r] = s;  // per-wave partial row sum
  }
  if (l16 == 0) {
    #pragma unroll
    for (int r = 0; r < 8; ++r) redbuf[r + 8 * half][w] = rowinv[r];
  }
  __syncthreads();
  #pragma unroll
  for (int r = 0; r < 8; ++r) {
    float s = 0.f;
    #pragma unroll
    for (int ww = 0; ww < 8; ++ww) s += redbuf[r + 8 * half][ww];
    rowinv[r] = 1.0f / s;
  }

  // normalize -> single NT f32 write to global attn + bf16 stripe in LDS
  float* aout = attn + (size_t)bh * SEQ * SEQ + (size_t)(mt * 16) * SEQ + w * 256;
  #pragma unroll
  for (int j = 0; j < 16; ++j) {
    #pragma unroll
    for (int r = 0; r < 8; ++r) {
      const float p = accS[j][r] * rowinv[r];
      const int row = r + 8 * half;
      const int col = j * 16 + l16;
      __builtin_nontemporal_store(p, &aout[(size_t)row * SEQ + col]);
      sP[row * SP_PITCH + w * 256 + col] = (__bf16)p;
    }
  }
  __syncthreads();

  // ---- phase 3: ctx partial = P[:, w*256 : w*256+256] @ V ----
  const __bf16* Bb = Vt + (size_t)bh * HD * SEQ;
  v8f accC[4] = {v8f{0}, v8f{0}, v8f{0}, v8f{0}};
  #pragma unroll 2
  for (int i = 0; i < 8; ++i) {
    const int kk = w * 256 + i * 32;
    v16bf a = loadA_bf16(sP + kk, SP_PITCH, lane);  // A-fragment straight from LDS
    #pragma unroll
    for (int j = 0; j < 4; ++j) {
      v16bf bf = loadB_k16(Bb + (size_t)(j * 16) * SEQ + kk, SEQ, lane);
      accC[j] = WMMA_BF16(a, bf, accC[j]);
    }
  }
  __syncthreads();                      // all waves done reading sP
  float* sPart = (float*)sP;            // overlay: 8 partial 16x64 f32 tiles (32KB)
  #pragma unroll
  for (int j = 0; j < 4; ++j)
    #pragma unroll
    for (int r = 0; r < 8; ++r)
      sPart[w * 1024 + (r + 8 * half) * 64 + j * 16 + l16] = accC[j][r];
  __syncthreads();

  // reduce the 8 partials -> bf16 ctx [B*S, 768]
  const int b = bh / NH, h = bh % NH;
  #pragma unroll
  for (int q = 0; q < 4; ++q) {
    const int idx = threadIdx.x + q * 256;  // 0..1023
    const int row = idx >> 6, col = idx & 63;
    float s = 0.f;
    #pragma unroll
    for (int ww = 0; ww < 8; ++ww) s += sPart[ww * 1024 + idx];
    ctx[(size_t)(b * SEQ + mt * 16 + row) * HIDW + h * 64 + col] = (__bf16)s;
  }
}

// ---------------- Kernel 4: out = ctx @ Wo + bo  (f32 output) ----------------
__global__ void outproj_kernel(const __bf16* __restrict__ ctx, const __bf16* __restrict__ WoT,
                               const float* __restrict__ bo, float* __restrict__ out) {
  const int wid  = (blockIdx.x * blockDim.x + threadIdx.x) >> 5;
  const int lane = threadIdx.x & 31;
  const int MT = (BATCH * SEQ) / 16;  // 512
  const int mt = wid % MT;
  const int ng = wid / MT;  // 0..11
  const __bf16* Ab = ctx + (size_t)mt * 16 * HIDW;

  v8f acc[4] = {v8f{0}, v8f{0}, v8f{0}, v8f{0}};
  for (int kk = 0; kk < HIDW; kk += 32) {
    v16bf a = loadA_bf16(Ab + kk, HIDW, lane);
    #pragma unroll
    for (int j = 0; j < 4; ++j) {
      v16bf bf = loadB_k16(WoT + (size_t)(ng * 64 + j * 16) * HIDW + kk, HIDW, lane);
      acc[j] = WMMA_BF16(a, bf, acc[j]);
    }
  }
  const int row0 = mt * 16 + ((lane >> 4) << 3);
  #pragma unroll
  for (int j = 0; j < 4; ++j) {
    const int n = ng * 64 + j * 16 + (lane & 15);
    const float bb = bo[n];
    #pragma unroll
    for (int r = 0; r < 8; ++r)
      out[(size_t)(row0 + r) * HIDW + n] = acc[j][r] + bb;
  }
}

// ---------------- Launch ----------------
extern "C" void kernel_launch(void* const* d_in, const int* in_sizes, int n_in,
                              void* d_out, int out_size, void* d_ws, size_t ws_size,
                              hipStream_t stream) {
  const float* x  = (const float*)d_in[0];
  const float* Wq = (const float*)d_in[1];
  const float* bq = (const float*)d_in[2];
  const float* Wk = (const float*)d_in[3];
  const float* bk = (const float*)d_in[4];
  const float* Wv = (const float*)d_in[5];
  const float* bv = (const float*)d_in[6];
  const float* Wo = (const float*)d_in[7];
  const float* bo = (const float*)d_in[8];

  float* out  = (float*)d_out;
  float* attn = out + (size_t)BATCH * SEQ * HIDW;  // second tuple output

  // Workspace layout (bf16 elements): 4x W^T | Q | K | V^T | ctx  ~= 55 MB
  __bf16* wsb = (__bf16*)d_ws;
  const size_t WTE = (size_t)HIDW * HIDW;            // 589824
  const size_t QKV = (size_t)BATCH * NH * SEQ * HD;  // 6291456
  __bf16* WqT = wsb;
  __bf16* WkT = wsb + WTE;
  __bf16* WvT = wsb + 2 * WTE;
  __bf16* WoT = wsb + 3 * WTE;
  __bf16* Qb  = wsb + 4 * WTE;
  __bf16* Kb  = Qb + QKV;
  __bf16* Vt  = Kb + QKV;
  __bf16* Cx  = Vt + QKV;

  // 1) weight transpose+convert: grid (24,24,4), block (32,8)
  w_transpose_kernel<<<dim3(24, 24, 4), dim3(32, 8), 0, stream>>>(Wq, Wk, Wv, Wo, WqT);

  // 2) QKV: 512 Mtiles * 12 Ngroups * 3 mats = 18432 waves -> 2304 blocks of 256
  qkv_gemm_kernel<<<2304, 256, 0, stream>>>(x, WqT, WkT, WvT, bq, bk, bv, Qb, Kb, Vt);

  // 3) fused attention: (B*H=48) * (S/16=128) blocks of 256 threads
  attn_fused_kernel<<<48 * 128, 256, 0, stream>>>(Qb, Kb, Vt, attn, Cx);

  // 4) out projection: 512 * 12 = 6144 waves -> 768 blocks
  outproj_kernel<<<768, 256, 0, stream>>>(Cx, WoT, bo, out);
}
